// MultiheadAttention_9423158247972
// MI455X (gfx1250) — matrix operand order repaired and checked by reading, not compiled, not run on hardware
//
#include <hip/hip_runtime.h>

// ---------------- problem constants ----------------
#define T_SEQ   2048
#define B_SZ    2
#define C_DIM   1024
#define H_NUM   16
#define D_HEAD  64
#define N_ROWS  (T_SEQ * B_SZ)          // 4096
#define BH      (B_SZ * H_NUM)          // 32 heads
#define SCALE_Q 0.125f                  // D_HEAD^-0.5
#define LOG2E   1.4426950408889634f

typedef __bf16 bf16_t;
typedef __attribute__((ext_vector_type(8)))  __bf16 v8bf;
typedef __attribute__((ext_vector_type(16))) __bf16 v16bf;
typedef __attribute__((ext_vector_type(8)))  float  v8f;

// Build a 16x32 (A) or 32x16 (B) bf16 fragment from two contiguous 16-byte chunks.
static __device__ __forceinline__ v16bf frag2(const bf16_t* p0, const bf16_t* p1) {
  const v8bf a = *(const v8bf*)p0;
  const v8bf b = *(const v8bf*)p1;
  v16bf r;
#pragma unroll
  for (int i = 0; i < 8; ++i) { r[i] = a[i]; r[i + 8] = b[i]; }
  return r;
}

static __device__ __forceinline__ v8f wmma_bf16(v16bf a, v16bf b, v8f c) {
  // D = A x B + C, 16x16x32, f32 accumulate
  return __builtin_amdgcn_wmma_f32_16x16x32_bf16(false, a, false, b, (short)0, c,
                                                 false, false);
}

// Async 16-byte copy global -> LDS (per lane), tracked with ASYNCcnt.
// LDS operand is the wave-relative LDS byte offset = low 32 bits of the flat addr.
static __device__ __forceinline__ void async_copy_b128(const bf16_t* gsrc,
                                                       bf16_t* lds_dst) {
  const unsigned loff = (unsigned)(size_t)lds_dst;
  const unsigned long long gaddr = (unsigned long long)(size_t)gsrc;
  asm volatile("global_load_async_to_lds_b128 %0, %1, off"
               :: "v"(loff), "v"(gaddr));
}

static __device__ __forceinline__ void wait_async0() {
  asm volatile("s_wait_asynccnt 0x0" ::: "memory");
}

// =====================================================================
// Kernel 1: fused QKV projection.  q/k/v = x2 @ W (q scaled).
// q,k written head-major (BH, T, D); v written head-major TRANSPOSED
// (BH, D, T) so the attention kernel can async-copy V tiles straight
// into its [dim][key] LDS layout.
// grid = (N/256, M/64, 3), block = 256 (8 waves, 2x4 wave grid, wave tile 32x64)
// =====================================================================
__global__ __launch_bounds__(256) void qkv_proj_kernel(
    const float* __restrict__ x,
    const float* __restrict__ wq, const float* __restrict__ wk,
    const float* __restrict__ wv,
    bf16_t* __restrict__ qb, bf16_t* __restrict__ kb, bf16_t* __restrict__ vb) {
  __shared__ bf16_t lds_a[64][40];    // x chunk, row-major [row][k], pad->40
  __shared__ bf16_t lds_bt[256][40];  // W chunk transposed [n][k], pad->40

  const float* w;  bf16_t* dst;  float scale;
  const int z = blockIdx.z;
  if (z == 0)      { w = wq; dst = qb; scale = SCALE_Q; }
  else if (z == 1) { w = wk; dst = kb; scale = 1.0f;    }
  else             { w = wv; dst = vb; scale = 1.0f;    }

  const int i0 = blockIdx.y * 64;
  const int j0 = blockIdx.x * 256;
  const int tid  = threadIdx.x;
  const int lane = tid & 31, wid = tid >> 5;
  const int wm = wid >> 2, wn = wid & 3;        // 2 x 4 waves
  const int l15 = lane & 15, lh = lane >> 4;

  const int arow = tid >> 3;                    // 0..31
  const int acol = (tid & 7) * 4;               // 0..28
  const int bk   = tid >> 6;                    // 0..3
  const int bn   = (tid & 63) * 4;              // 0..252

  v8f acc[2][4];
#pragma unroll
  for (int mt = 0; mt < 2; ++mt)
#pragma unroll
    for (int nt = 0; nt < 4; ++nt)
#pragma unroll
      for (int g = 0; g < 8; ++g) acc[mt][nt][g] = 0.0f;

  for (int k0 = 0; k0 < C_DIM; k0 += 32) {
    __syncthreads();
    // ---- stage A (x) fp32 -> bf16 ----
#pragma unroll
    for (int it = 0; it < 2; ++it) {
      const int r = arow + 32 * it;
      const float4 f = *(const float4*)(x + (size_t)(i0 + r) * C_DIM + k0 + acol);
      lds_a[r][acol + 0] = (bf16_t)f.x;
      lds_a[r][acol + 1] = (bf16_t)f.y;
      lds_a[r][acol + 2] = (bf16_t)f.z;
      lds_a[r][acol + 3] = (bf16_t)f.w;
    }
    // ---- stage B (W) fp32 -> bf16, transposed [n][k] ----
#pragma unroll
    for (int it = 0; it < 8; ++it) {
      const int k = bk + 4 * it;
      const float4 f = *(const float4*)(w + (size_t)(k0 + k) * C_DIM + j0 + bn);
      lds_bt[bn + 0][k] = (bf16_t)f.x;
      lds_bt[bn + 1][k] = (bf16_t)f.y;
      lds_bt[bn + 2][k] = (bf16_t)f.z;
      lds_bt[bn + 3][k] = (bf16_t)f.w;
    }
    if (k0 + 32 < C_DIM) {
      __builtin_prefetch(x + (size_t)(i0 + arow) * C_DIM + k0 + 32 + acol, 0, 1);
      __builtin_prefetch(w + (size_t)(k0 + 32 + bk) * C_DIM + j0 + bn, 0, 1);
    }
    __syncthreads();

    v16bf af[2], bfr[4];
#pragma unroll
    for (int mt = 0; mt < 2; ++mt) {
      const bf16_t* ap = &lds_a[32 * wm + 16 * mt + l15][0];
      af[mt] = frag2(ap + 8 * lh, ap + 16 + 8 * lh);
    }
#pragma unroll
    for (int nt = 0; nt < 4; ++nt) {
      const bf16_t* bp = &lds_bt[64 * wn + 16 * nt + l15][0];
      bfr[nt] = frag2(bp + 8 * lh, bp + 16 + 8 * lh);
    }
#pragma unroll
    for (int mt = 0; mt < 2; ++mt)
#pragma unroll
      for (int nt = 0; nt < 4; ++nt)
        acc[mt][nt] = wmma_bf16(af[mt], bfr[nt], acc[mt][nt]);
  }

  // ---- store epilogue: uniform branch hoisted out of the unrolled loops ----
  if (z != 2) {
    // q/k -> (BH, T, D)
#pragma unroll
    for (int mt = 0; mt < 2; ++mt)
#pragma unroll
      for (int nt = 0; nt < 4; ++nt)
#pragma unroll
        for (int g = 0; g < 8; ++g) {
          const int i = i0 + 32 * wm + 16 * mt + g + 8 * lh;   // row of x2
          const int j = j0 + 64 * wn + 16 * nt + l15;          // channel
          const int head = (i & 1) * H_NUM + (j >> 6);
          dst[(size_t)head * (T_SEQ * D_HEAD) + (size_t)(i >> 1) * D_HEAD +
              (j & 63)] = (bf16_t)(acc[mt][nt][g] * scale);
        }
  } else {
    // v -> (BH, D, T)
#pragma unroll
    for (int mt = 0; mt < 2; ++mt)
#pragma unroll
      for (int nt = 0; nt < 4; ++nt)
#pragma unroll
        for (int g = 0; g < 8; ++g) {
          const int i = i0 + 32 * wm + 16 * mt + g + 8 * lh;
          const int j = j0 + 64 * wn + 16 * nt + l15;
          const int head = (i & 1) * H_NUM + (j >> 6);
          dst[(size_t)head * (T_SEQ * D_HEAD) + (size_t)(j & 63) * T_SEQ +
              (i >> 1)] = (bf16_t)acc[mt][nt][g];
        }
  }
}

// =====================================================================
// Kernel 2: flash attention per head.  grid = (BH, T/128), block = 256.
// Each wave owns a 16-row query tile.  K tile ([key][dim]) and V tile
// ([dim][key], already transposed in global memory) are async-copied
// into LDS via GLOBAL_LOAD_ASYNC_TO_LDS_B128 (ASYNCcnt).
// =====================================================================
__global__ __launch_bounds__(256) void flash_attn_kernel(
    const bf16_t* __restrict__ qb, const bf16_t* __restrict__ kb,
    const bf16_t* __restrict__ vb, bf16_t* __restrict__ ctx) {
  __shared__ bf16_t kt[64][72];      // [key][dim]
  __shared__ bf16_t vt[64][72];      // [dim][key]
  __shared__ bf16_t ps[8][16][72];   // per-wave P re-swizzle scratch

  const int head = blockIdx.x;
  const int q0   = blockIdx.y * 128;
  const int tid  = threadIdx.x;
  const int lane = tid & 31, wv_ = tid >> 5;
  const int l15 = lane & 15, lh = lane >> 4;

  const bf16_t* qh  = qb + (size_t)head * T_SEQ * D_HEAD;
  const bf16_t* kh  = kb + (size_t)head * T_SEQ * D_HEAD;
  const bf16_t* vhT = vb + (size_t)head * T_SEQ * D_HEAD;  // [d][t]

  // Q fragments: 16 rows x 64 dims -> 2 A-frags (K-chunks of 32)
  const bf16_t* qp = qh + (size_t)(q0 + 16 * wv_ + l15) * D_HEAD;
  v16bf qf[2];
  qf[0] = frag2(qp + 8 * lh,      qp + 16 + 8 * lh);
  qf[1] = frag2(qp + 32 + 8 * lh, qp + 48 + 8 * lh);

  // per-lane staging coordinates (element index within a 64x64 tile)
  const int e0 = tid * 8;
  const int r0 = e0 >> 6, c0 = e0 & 63;        // chunk 0: rows 0..31
  const int r1 = r0 + 32;                      // chunk 1: rows 32..63

  float m[8], lsum[8];
  v8f o[4];
#pragma unroll
  for (int g = 0; g < 8; ++g) { m[g] = -1e30f; lsum[g] = 0.0f; }
#pragma unroll
  for (int nt = 0; nt < 4; ++nt)
#pragma unroll
    for (int g = 0; g < 8; ++g) o[nt][g] = 0.0f;

  for (int kt0 = 0; kt0 < T_SEQ; kt0 += 64) {
    __syncthreads();  // previous tile's LDS consumers are done
    // ---- async-stage K tile and V tile: 16B per lane per chunk ----
    async_copy_b128(kh + (size_t)(kt0 + r0) * D_HEAD + c0, &kt[r0][c0]);
    async_copy_b128(kh + (size_t)(kt0 + r1) * D_HEAD + c0, &kt[r1][c0]);
    async_copy_b128(vhT + (size_t)r0 * T_SEQ + kt0 + c0, &vt[r0][c0]);
    async_copy_b128(vhT + (size_t)r1 * T_SEQ + kt0 + c0, &vt[r1][c0]);
    wait_async0();
    __syncthreads();

    // ---- S = Q @ K^T : 4 key tiles x 2 K-chunks ----
    v8f s[4];
#pragma unroll
    for (int nt = 0; nt < 4; ++nt) {
#pragma unroll
      for (int g = 0; g < 8; ++g) s[nt][g] = 0.0f;
      const bf16_t* kp = &kt[16 * nt + l15][0];
#pragma unroll
      for (int kc = 0; kc < 2; ++kc) {
        const v16bf bfrag = frag2(kp + 32 * kc + 8 * lh, kp + 32 * kc + 16 + 8 * lh);
        s[nt] = wmma_bf16(qf[kc], bfrag, s[nt]);
      }
    }

    // ---- online softmax: row max / rescale / exp / row sum ----
    float rmax[8], rsum[8], cf[8];
#pragma unroll
    for (int g = 0; g < 8; ++g) {
      float v = s[0][g];
#pragma unroll
      for (int nt = 1; nt < 4; ++nt) v = fmaxf(v, s[nt][g]);
      rmax[g] = v;
    }
#pragma unroll
    for (int mask = 1; mask < 16; mask <<= 1)
#pragma unroll
      for (int g = 0; g < 8; ++g)
        rmax[g] = fmaxf(rmax[g], __shfl_xor(rmax[g], mask, 16));
#pragma unroll
    for (int g = 0; g < 8; ++g) {
      const float mn = fmaxf(m[g], rmax[g]);
      cf[g] = exp2f((m[g] - mn) * LOG2E);
      m[g] = mn;
      rsum[g] = 0.0f;
    }
#pragma unroll
    for (int nt = 0; nt < 4; ++nt)
#pragma unroll
      for (int g = 0; g < 8; ++g) {
        const float p = exp2f((s[nt][g] - m[g]) * LOG2E);
        rsum[g] += p;
        ps[wv_][g + 8 * lh][16 * nt + l15] = (bf16_t)p;  // C-layout -> row-major
      }
#pragma unroll
    for (int mask = 1; mask < 16; mask <<= 1)
#pragma unroll
      for (int g = 0; g < 8; ++g)
        rsum[g] += __shfl_xor(rsum[g], mask, 16);
#pragma unroll
    for (int g = 0; g < 8; ++g) lsum[g] = lsum[g] * cf[g] + rsum[g];
#pragma unroll
    for (int nt = 0; nt < 4; ++nt)
#pragma unroll
      for (int g = 0; g < 8; ++g) o[nt][g] *= cf[g];

    // ---- O += P @ V : P re-read from own-wave LDS in A-frag layout ----
    const bf16_t* pp = &ps[wv_][l15][0];
    v16bf pf[2];
#pragma unroll
    for (int kc = 0; kc < 2; ++kc)
      pf[kc] = frag2(pp + 32 * kc + 8 * lh, pp + 32 * kc + 16 + 8 * lh);
#pragma unroll
    for (int nt = 0; nt < 4; ++nt) {
      const bf16_t* vp = &vt[16 * nt + l15][0];
#pragma unroll
      for (int kc = 0; kc < 2; ++kc) {
        const v16bf vfrag = frag2(vp + 32 * kc + 8 * lh, vp + 32 * kc + 16 + 8 * lh);
        o[nt] = wmma_bf16(pf[kc], vfrag, o[nt]);
      }
    }
  }

  // ---- normalize and store ctx back in (T*B, C) row-major bf16 ----
  const int b_ = head >> 4, h = head & 15;
#pragma unroll
  for (int g = 0; g < 8; ++g) {
    const float rinv = 1.0f / lsum[g];
    const int t = q0 + 16 * wv_ + g + 8 * lh;
#pragma unroll
    for (int nt = 0; nt < 4; ++nt) {
      const int d = 16 * nt + l15;
      ctx[(size_t)(t * B_SZ + b_) * C_DIM + h * D_HEAD + d] =
          (bf16_t)(o[nt][g] * rinv);
    }
  }
}

// =====================================================================
// Kernel 3: output projection out = ctx @ wo + bo (fp32 out).
// A tile (bf16 ctx) is async-copied into LDS.
// grid = (N/256, M/64), block = 256.
// =====================================================================
__global__ __launch_bounds__(256) void out_proj_kernel(
    const bf16_t* __restrict__ ctx, const float* __restrict__ wo,
    const float* __restrict__ bo, float* __restrict__ out) {
  __shared__ bf16_t lds_a[64][40];
  __shared__ bf16_t lds_bt[256][40];

  const int i0 = blockIdx.y * 64;
  const int j0 = blockIdx.x * 256;
  const int tid  = threadIdx.x;
  const int lane = tid & 31, wid = tid >> 5;
  const int wm = wid >> 2, wn = wid & 3;
  const int l15 = lane & 15, lh = lane >> 4;

  const int arow = tid >> 2;          // 0..63
  const int acol = (tid & 3) * 8;     // 0,8,16,24
  const int bk   = tid >> 6;
  const int bn   = (tid & 63) * 4;

  v8f acc[2][4];
#pragma unroll
  for (int mt = 0; mt < 2; ++mt)
#pragma unroll
    for (int nt = 0; nt < 4; ++nt)
#pragma unroll
      for (int g = 0; g < 8; ++g) acc[mt][nt][g] = 0.0f;

  for (int k0 = 0; k0 < C_DIM; k0 += 32) {
    __syncthreads();
    // ---- A tile: bf16 -> bf16 straight async copy ----
    async_copy_b128(ctx + (size_t)(i0 + arow) * C_DIM + k0 + acol,
                    &lds_a[arow][acol]);
    // ---- B tile: fp32 -> bf16 transposed [n][k] ----
#pragma unroll
    for (int it = 0; it < 8; ++it) {
      const int k = bk + 4 * it;
      const float4 f = *(const float4*)(wo + (size_t)(k0 + k) * C_DIM + j0 + bn);
      lds_bt[bn + 0][k] = (bf16_t)f.x;
      lds_bt[bn + 1][k] = (bf16_t)f.y;
      lds_bt[bn + 2][k] = (bf16_t)f.z;
      lds_bt[bn + 3][k] = (bf16_t)f.w;
    }
    if (k0 + 32 < C_DIM)
      __builtin_prefetch(wo + (size_t)(k0 + 32 + bk) * C_DIM + j0 + bn, 0, 1);
    wait_async0();
    __syncthreads();

    v16bf af[2], bfr[4];
#pragma unroll
    for (int mt = 0; mt < 2; ++mt) {
      const bf16_t* ap = &lds_a[32 * wm + 16 * mt + l15][0];
      af[mt] = frag2(ap + 8 * lh, ap + 16 + 8 * lh);
    }
#pragma unroll
    for (int nt = 0; nt < 4; ++nt) {
      const bf16_t* bp = &lds_bt[64 * wn + 16 * nt + l15][0];
      bfr[nt] = frag2(bp + 8 * lh, bp + 16 + 8 * lh);
    }
#pragma unroll
    for (int mt = 0; mt < 2; ++mt)
#pragma unroll
      for (int nt = 0; nt < 4; ++nt)
        acc[mt][nt] = wmma_bf16(af[mt], bfr[nt], acc[mt][nt]);
  }

  float bias[4];
#pragma unroll
  for (int nt = 0; nt < 4; ++nt) bias[nt] = bo[j0 + 64 * wn + 16 * nt + l15];

#pragma unroll
  for (int mt = 0; mt < 2; ++mt)
#pragma unroll
    for (int nt = 0; nt < 4; ++nt)
#pragma unroll
      for (int g = 0; g < 8; ++g) {
        const int i = i0 + 32 * wm + 16 * mt + g + 8 * lh;
        const int j = j0 + 64 * wn + 16 * nt + l15;
        out[(size_t)i * C_DIM + j] = acc[mt][nt][g] + bias[nt];
      }
}

// =====================================================================
extern "C" void kernel_launch(void* const* d_in, const int* in_sizes, int n_in,
                              void* d_out, int out_size, void* d_ws, size_t ws_size,
                              hipStream_t stream) {
  const float* x  = (const float*)d_in[0];
  const float* wq = (const float*)d_in[1];
  const float* wk = (const float*)d_in[2];
  const float* wv = (const float*)d_in[3];
  const float* wo = (const float*)d_in[4];
  const float* bo = (const float*)d_in[5];
  float* out = (float*)d_out;

  const size_t headsz = (size_t)BH * T_SEQ * D_HEAD;  // 4M bf16 elems
  bf16_t* qb = (bf16_t*)d_ws;
  bf16_t* kb = qb + headsz;
  bf16_t* vb = kb + headsz;   // stored (BH, D, T)
  bf16_t* cx = vb + headsz;   // (N_ROWS, C) bf16

  dim3 blk(256);
  qkv_proj_kernel<<<dim3(C_DIM / 256, N_ROWS / 64, 3), blk, 0, stream>>>(
      x, wq, wk, wv, qb, kb, vb);
  flash_attn_kernel<<<dim3(BH, T_SEQ / 128), blk, 0, stream>>>(qb, kb, vb, cx);
  out_proj_kernel<<<dim3(C_DIM / 256, N_ROWS / 64), blk, 0, stream>>>(
      cx, wo, bo, out);
}
